// NystromSelfAttention_35167192220064
// MI455X (gfx1250) — matrix operand order repaired and checked by reading, not compiled, not run on hardware
//
#include <hip/hip_runtime.h>

// ---------------------------------------------------------------------------
// Nystrom self-attention for MI455X (gfx1250), built around v_wmma_f32_16x16x32_f16.
// pinv(kernel_2) replaced by Newton-Schulz iteration Z <- Z(2I - A Z),
// seeded Z0 = A^T / (norm1(A) * normInf(A)); normInf == 1 for softmax rows.
//
// GEMM: 128x64x32 block tile, 8 waves (4x2), each wave computes 2x2 16x16
// accumulators. Tiles are staged to LDS pre-permuted into the wave32 WMMA
// fragment layout, so each operand is a single 32B contiguous LDS read.
// ---------------------------------------------------------------------------

typedef _Float16 f16;
typedef __attribute__((ext_vector_type(16))) _Float16 v16h;
typedef __attribute__((ext_vector_type(4)))  _Float16 v4h;
typedef __attribute__((ext_vector_type(8)))  float    v8f;

#define BM 128
#define BN 64
#define BK 32
#define MLAND 64
#define SEG 16

// C (M x N) = alpha * (A(MxK) @ op(B) + bias)   [batched via blockIdx.z]
// TRANSB=1: B is (N x K) row-major (contract last dims);  TRANSB=0: B is (K x N).
// Requires: M % 128 == 0, N % 64 == 0, K % 32 == 0 (true for all call sites).
template<int TRANSB, int HAS_BIAS>
__global__ __launch_bounds__(256)
void gemm_wmma(const float* __restrict__ A, const float* __restrict__ Bm,
               const float* __restrict__ bias, float* __restrict__ C,
               int Mdim, int Ndim, int Kdim,
               long long strA, long long strB, long long strC, float alpha)
{
    // Fragment-ordered LDS: [16x16-tile][lane][16 halves] -> 32B/lane contiguous.
    __shared__ f16 Apack[BM / 16][32][16];   //  8 KB
    __shared__ f16 Bpack[BN / 16][32][16];   //  4 KB

    const int b = blockIdx.z;
    A  += (long long)b * strA;
    Bm += (long long)b * strB;
    C  += (long long)b * strC;

    const int lane = threadIdx.x;            // 0..31 (wave32)
    const int wave = threadIdx.y;            // 0..7
    const int tid  = wave * 32 + lane;
    const int wy   = wave >> 1;              // 0..3 along M (2 tiles each)
    const int wx   = wave & 1;               // 0..1 along N (2 tiles each)
    const int rowBase = blockIdx.y * BM;
    const int colBase = blockIdx.x * BN;
    const int ml = lane & 15;
    const int hh = lane >> 4;

    v8f acc[2][2] = {};

    for (int k0 = 0; k0 < Kdim; k0 += BK) {
        if (k0 + BK < Kdim)
            __builtin_prefetch(A + (long long)(rowBase + (tid >> 1)) * Kdim + (k0 + BK), 0, 1);

        // ---- Stage A: 128x32 f32 as float4 along K (1024 vec4, 4/thread) ----
        // A frag layout: lane = (m&15) + 16*(k>>4), elem = k&15.
        #pragma unroll
        for (int i = 0; i < 4; ++i) {
            const int vi = tid + 256 * i;
            const int r  = vi >> 3;                 // 0..127
            const int kq = (vi & 7) << 2;           // 0,4,..,28
            const float4 f = *(const float4*)(A + (long long)(rowBase + r) * Kdim + (k0 + kq));
            const int ln = (r & 15) + ((kq >> 4) << 4);
            const int eb = kq & 15;
            v4h h4 = { (f16)f.x, (f16)f.y, (f16)f.z, (f16)f.w };
            *(v4h*)&Apack[r >> 4][ln][eb] = h4;
        }
        // ---- Stage B: 64x32 (B frag layout: lane = (n&15) + 16*(k>>4), elem = k&15) ----
        if (TRANSB) {
            #pragma unroll
            for (int i = 0; i < 2; ++i) {
                const int vi = tid + 256 * i;       // 512 vec4
                const int n  = vi >> 3;             // 0..63
                const int kq = (vi & 7) << 2;
                const float4 f = *(const float4*)(Bm + (long long)(colBase + n) * Kdim + (k0 + kq));
                const int ln = (n & 15) + (((kq >> 4) & 1) << 4);
                const int eb = kq & 15;
                v4h h4 = { (f16)f.x, (f16)f.y, (f16)f.z, (f16)f.w };
                *(v4h*)&Bpack[n >> 4][ln][eb] = h4;
            }
        } else {
            #pragma unroll
            for (int i = 0; i < 2; ++i) {
                const int ui = tid + 256 * i;       // 512 units: 4 K-elems @ fixed n
                const int n  = ui & 63;
                const int kq = (ui >> 6) << 2;      // 0,4,..,28
                float x0 = Bm[(long long)(k0 + kq + 0) * Ndim + (colBase + n)];
                float x1 = Bm[(long long)(k0 + kq + 1) * Ndim + (colBase + n)];
                float x2 = Bm[(long long)(k0 + kq + 2) * Ndim + (colBase + n)];
                float x3 = Bm[(long long)(k0 + kq + 3) * Ndim + (colBase + n)];
                const int ln = (n & 15) + (((kq >> 4) & 1) << 4);
                const int eb = kq & 15;
                v4h h4 = { (f16)x0, (f16)x1, (f16)x2, (f16)x3 };
                *(v4h*)&Bpack[n >> 4][ln][eb] = h4;
            }
        }
        __syncthreads();

        // ---- Fragments: one 32B contiguous LDS read each (2x ds_load_b128) ----
        v16h af[2], bf[2];
        #pragma unroll
        for (int i = 0; i < 2; ++i) af[i] = *(const v16h*)&Apack[wy * 2 + i][lane][0];
        #pragma unroll
        for (int j = 0; j < 2; ++j) bf[j] = *(const v16h*)&Bpack[wx * 2 + j][lane][0];

        #pragma unroll
        for (int i = 0; i < 2; ++i)
            #pragma unroll
            for (int j = 0; j < 2; ++j)
                acc[i][j] = __builtin_amdgcn_wmma_f32_16x16x32_f16(
                    false, af[i], false, bf[j], (short)0, acc[i][j], false, false);
        __syncthreads();
    }

    // ---- Epilogue: C/D layout M = v + 8*h, N = lane&15 ----
    #pragma unroll
    for (int j = 0; j < 2; ++j) {
        const int col = colBase + (wx * 2 + j) * 16 + ml;
        const float bvv = HAS_BIAS ? bias[col] : 0.0f;
        #pragma unroll
        for (int i = 0; i < 2; ++i) {
            #pragma unroll
            for (int v = 0; v < 8; ++v) {
                const int row = rowBase + (wy * 2 + i) * 16 + v + (hh << 3);
                C[(long long)row * Ndim + col] = alpha * (acc[i][j][v] + bvv);
            }
        }
    }
}

// Row-wise softmax; one block (256 threads) per row, float4 traffic.
__global__ __launch_bounds__(256)
void softmax_rows(float* __restrict__ X, int Ndim)
{
    __shared__ float red[256];
    float4* p = (float4*)(X + (long long)blockIdx.x * Ndim);
    const int n4 = Ndim >> 2;
    const int tid = threadIdx.x;

    float mx = -1e30f;
    for (int j = tid; j < n4; j += 256) {
        float4 f = p[j];
        mx = fmaxf(mx, fmaxf(fmaxf(f.x, f.y), fmaxf(f.z, f.w)));
    }
    red[tid] = mx; __syncthreads();
    for (int s = 128; s > 0; s >>= 1) { if (tid < s) red[tid] = fmaxf(red[tid], red[tid + s]); __syncthreads(); }
    mx = red[0]; __syncthreads();

    float sum = 0.0f;
    for (int j = tid; j < n4; j += 256) {
        float4 f = p[j];
        f.x = __expf(f.x - mx); f.y = __expf(f.y - mx);
        f.z = __expf(f.z - mx); f.w = __expf(f.w - mx);
        p[j] = f;
        sum += f.x + f.y + f.z + f.w;
    }
    red[tid] = sum; __syncthreads();
    for (int s = 128; s > 0; s >>= 1) { if (tid < s) red[tid] += red[tid + s]; __syncthreads(); }
    const float inv = 1.0f / red[0];
    for (int j = tid; j < n4; j += 256) {
        float4 f = p[j];
        f.x *= inv; f.y *= inv; f.z *= inv; f.w *= inv;
        p[j] = f;
    }
}

// Landmark selection: rank MLAND segment-blocks of k by squared norm,
// gather in descending-score order into kl. One block per batch.
__global__ __launch_bounds__(256)
void landmarks_kernel(const float* __restrict__ k, float* __restrict__ kl,
                      int S, int DKd)
{
    __shared__ float sc[MLAND];
    __shared__ int   idx[MLAND];
    __shared__ float red[256];
    const int b = blockIdx.x;
    const float* kb  = k  + (long long)b * S * DKd;
    float*       klb = kl + (long long)b * S * DKd;
    const int tid = threadIdx.x;

    for (int mm = 0; mm < MLAND; ++mm) {
        float s = 0.0f;
        const float4* sp = (const float4*)(kb + (long long)mm * SEG * DKd);
        const int n4 = (SEG * DKd) >> 2;
        for (int e = tid; e < n4; e += 256) {
            float4 f = sp[e];
            s += f.x * f.x + f.y * f.y + f.z * f.z + f.w * f.w;
        }
        red[tid] = s; __syncthreads();
        for (int st = 128; st > 0; st >>= 1) { if (tid < st) red[tid] += red[tid + st]; __syncthreads(); }
        if (tid == 0) sc[mm] = red[0];
        __syncthreads();
    }
    if (tid == 0) {                                  // descending selection sort
        for (int i = 0; i < MLAND; ++i) idx[i] = i;
        for (int i = 0; i < MLAND; ++i) {
            int best = i;
            for (int j = i + 1; j < MLAND; ++j) if (sc[idx[j]] > sc[idx[best]]) best = j;
            int t = idx[i]; idx[i] = idx[best]; idx[best] = t;
        }
    }
    __syncthreads();
    for (int mo = 0; mo < MLAND; ++mo) {
        const float4* sp = (const float4*)(kb  + (long long)idx[mo] * SEG * DKd);
        float4*       dp = (float4*)      (klb + (long long)mo      * SEG * DKd);
        const int n4 = (SEG * DKd) >> 2;
        for (int e = tid; e < n4; e += 256) dp[e] = sp[e];
    }
}

// colsum[b][j] = sum_i |A[b][i][j]|
__global__ __launch_bounds__(256)
void col_abs_sum(const float* __restrict__ A, float* __restrict__ colsum, int S)
{
    const int b = blockIdx.z;
    const int j = blockIdx.x * 256 + threadIdx.x;
    const float* Ab = A + (long long)b * S * S;
    float s = 0.0f;
    for (int i = 0; i < S; ++i) s += fabsf(Ab[(long long)i * S + j]);
    colsum[b * S + j] = s;
}

// scale[b] = 1 / max_j colsum[b][j]   (normInf == 1 for softmax rows)
__global__ __launch_bounds__(256)
void scale_from_colsum(const float* __restrict__ colsum, float* __restrict__ scale, int S)
{
    __shared__ float red[256];
    const int b = blockIdx.x, tid = threadIdx.x;
    float mx = 0.0f;
    for (int j = tid; j < S; j += 256) mx = fmaxf(mx, colsum[b * S + j]);
    red[tid] = mx; __syncthreads();
    for (int s = 128; s > 0; s >>= 1) { if (tid < s) red[tid] = fmaxf(red[tid], red[tid + s]); __syncthreads(); }
    if (tid == 0) scale[b] = 1.0f / red[0];
}

// Z[b] = scale[b] * A[b]^T   (32x32 LDS transpose tiles)
__global__ __launch_bounds__(256)
void transpose_scale(const float* __restrict__ A, const float* __restrict__ scale,
                     float* __restrict__ Z, int S)
{
    __shared__ float t[32][33];
    const int b = blockIdx.z;
    const float* Ab = A + (long long)b * S * S;
    float*       Zb = Z + (long long)b * S * S;
    const int x0 = blockIdx.x * 32, y0 = blockIdx.y * 32;
    const int tx = threadIdx.x, ty = threadIdx.y;        // (32, 8)
    for (int i = ty; i < 32; i += 8) t[i][tx] = Ab[(long long)(y0 + i) * S + (x0 + tx)];
    __syncthreads();
    const float sc = scale[b];
    for (int i = ty; i < 32; i += 8) Zb[(long long)(x0 + i) * S + (y0 + tx)] = sc * t[tx][i];
}

// W = 2I - P   (safe in place)
__global__ __launch_bounds__(256)
void two_i_minus(const float* __restrict__ P, float* __restrict__ W, int S)
{
    const int b = blockIdx.z, i = blockIdx.y;
    const int j = blockIdx.x * 256 + threadIdx.x;
    const long long o = (long long)b * S * S + (long long)i * S + j;
    float v = -P[o];
    if (i == j) v += 2.0f;
    W[o] = v;
}

extern "C" void kernel_launch(void* const* d_in, const int* in_sizes, int n_in,
                              void* d_out, int out_size, void* d_ws, size_t ws_size,
                              hipStream_t stream)
{
    (void)in_sizes; (void)n_in; (void)out_size; (void)ws_size;

    const float* query = (const float*)d_in[0];
    const float* key   = (const float*)d_in[1];
    const float* value = (const float*)d_in[2];
    const float* Wq    = (const float*)d_in[3];
    const float* bq    = (const float*)d_in[4];
    const float* Wk    = (const float*)d_in[5];
    const float* bk    = (const float*)d_in[6];
    const float* Wv    = (const float*)d_in[7];
    const float* bv    = (const float*)d_in[8];
    float* out = (float*)d_out;

    const int Bb = 8, S = 1024, E = 1024, DKd = 512, DVd = 512;
    const long long PROJ = (long long)Bb * S * DKd;      // 4M floats
    const long long MAT  = (long long)Bb * S * S;        // 8M floats
    const long long MATS = (long long)S * S;
    const long long PRS  = (long long)S * DKd;

    float* ws = (float*)d_ws;
    float* q      = ws;                 // PROJ
    float* k      = q  + PROJ;          // PROJ
    float* v      = k  + PROJ;          // PROJ
    float* kl     = v  + PROJ;          // PROJ
    float* K1     = kl + PROJ;          // MAT  (kernel_1 == kernel_2)
    float* K3     = K1 + MAT;           // MAT
    float* Z      = K3 + MAT;           // MAT
    float* P      = Z  + MAT;           // MAT  (AZ scratch, later T2)
    float* Z2     = P  + MAT;           // MAT
    float* T1     = Z2 + MAT;           // PROJ (kernel_3 @ v)
    float* colsum = T1 + PROJ;          // Bb*S
    float* scale  = colsum + (long long)Bb * S;  // Bb

    const dim3 blk(32, 8);
    const float sc = 0.21022410381342863f;       // 512^(-1/4)

    // ---- Projections: (X @ W^T + b) [* scale for q,k] ----
    const dim3 gproj(DKd / BN, (Bb * S) / BM, 1);
    gemm_wmma<1, 1><<<gproj, blk, 0, stream>>>(query, Wq, bq, q, Bb * S, DKd, E, 0, 0, 0, sc);
    gemm_wmma<1, 1><<<gproj, blk, 0, stream>>>(key,   Wk, bk, k, Bb * S, DKd, E, 0, 0, 0, sc);
    gemm_wmma<1, 1><<<gproj, blk, 0, stream>>>(value, Wv, bv, v, Bb * S, DVd, E, 0, 0, 0, 1.0f);

    // ---- Landmark selection ----
    landmarks_kernel<<<Bb, 256, 0, stream>>>(k, kl, S, DKd);

    // ---- kernel_1 = softmax(q @ kl^T), kernel_3 = softmax(q @ k^T) ----
    const dim3 gscore(S / BN, S / BM, Bb);
    gemm_wmma<1, 0><<<gscore, blk, 0, stream>>>(q, kl, nullptr, K1, S, S, DKd, PRS, PRS, MATS, 1.0f);
    gemm_wmma<1, 0><<<gscore, blk, 0, stream>>>(q, k,  nullptr, K3, S, S, DKd, PRS, PRS, MATS, 1.0f);
    softmax_rows<<<Bb * S, 256, 0, stream>>>(K1, S);
    softmax_rows<<<Bb * S, 256, 0, stream>>>(K3, S);

    // ---- Newton-Schulz pinv seed: Z0 = A^T / norm1(A) ----
    col_abs_sum<<<dim3(S / 256, 1, Bb), 256, 0, stream>>>(K1, colsum, S);
    scale_from_colsum<<<Bb, 256, 0, stream>>>(colsum, scale, S);
    transpose_scale<<<dim3(S / 32, S / 32, Bb), dim3(32, 8), 0, stream>>>(K1, scale, Z, S);

    // ---- Newton-Schulz iterations: Z <- Z (2I - A Z) ----
    float* Za = Z;
    float* Zn = Z2;
    const dim3 gsq(S / BN, S / BM, Bb);
    for (int it = 0; it < 10; ++it) {
        gemm_wmma<0, 0><<<gsq, blk, 0, stream>>>(K1, Za, nullptr, P, S, S, S, MATS, MATS, MATS, 1.0f);
        two_i_minus<<<dim3(S / 256, S, Bb), 256, 0, stream>>>(P, P, S);
        gemm_wmma<0, 0><<<gsq, blk, 0, stream>>>(Za, P, nullptr, Zn, S, S, S, MATS, MATS, MATS, 1.0f);
        float* t = Za; Za = Zn; Zn = t;
    }

    // ---- out = (K1 @ Z) @ (K3 @ v) ----
    const dim3 gnv(DVd / BN, S / BM, Bb);
    gemm_wmma<0, 0><<<gnv, blk, 0, stream>>>(K3, v, nullptr, T1, S, DVd, S, MATS, PRS, PRS, 1.0f);
    gemm_wmma<0, 0><<<gsq, blk, 0, stream>>>(K1, Za, nullptr, P, S, S, S, MATS, MATS, MATS, 1.0f);
    gemm_wmma<0, 0><<<gnv, blk, 0, stream>>>(P, T1, nullptr, out, S, DVd, S, MATS, PRS, PRS, 1.0f);

    (void)scale;
}
